// InteractionBlock_55645596287232
// MI455X (gfx1250) — matrix operand order repaired and checked by reading, not compiled, not run on hardware
//
#include <hip/hip_runtime.h>
#include <hip/hip_bf16.h>

#define NNODES 100000
#define NEDGES 1600000
#define NREL 4
#define HDIM 128
#define ADIM 64
#define MIDDIM 64

typedef __attribute__((ext_vector_type(16))) __bf16 bf16x16;
typedef __attribute__((ext_vector_type(8)))  float  f32x8;
typedef __attribute__((ext_vector_type(4)))  unsigned int u32x4;
typedef __attribute__((ext_vector_type(8)))  int i32x8;
typedef __attribute__((ext_vector_type(4)))  int i32x4;

// ---------- bf16 helpers (RNE) ----------
__device__ __forceinline__ unsigned short f32_bf16_bits(float f) {
  union { float f; unsigned int u; } c; c.f = f;
  unsigned int u = c.u + (0x7FFFu + ((c.u >> 16) & 1u));
  return (unsigned short)(u >> 16);
}
__device__ __forceinline__ float bf16lo_f32(unsigned int packed) {
  union { unsigned int u; float f; } c; c.u = packed << 16; return c.f;
}
__device__ __forceinline__ float bf16hi_f32(unsigned int packed) {
  union { unsigned int u; float f; } c; c.u = packed & 0xFFFF0000u; return c.f;
}
__device__ __forceinline__ unsigned int pack_bf16x2(float lo, float hi) {
  union { float f; unsigned int u; } a, b;
  a.f = lo; b.f = hi;
  unsigned int ua = a.u + (0x7FFFu + ((a.u >> 16) & 1u));
  unsigned int ub = b.u + (0x7FFFu + ((b.u >> 16) & 1u));
  return (ua >> 16) | (ub & 0xFFFF0000u);
}

// ---------- TDM: 2D bf16 tile (32 elems wide) global -> LDS, padded to 40-elem rows ----------
// lds_off: LDS byte address of tile start. gptr: bf16 tile start in global.
// tileRows: rows DMAed. validRows: rows valid in memory (beyond -> hardware zero-fill).
// strideElems: row stride of source tensor in elements.
__device__ __forceinline__ void tdm_load_tile_bf16(unsigned lds_off,
                                                   const unsigned short* gptr,
                                                   unsigned tileRows,
                                                   unsigned validRows,
                                                   unsigned strideElems) {
  unsigned long long ga = (unsigned long long)(size_t)gptr;
  u32x4 g0;
  g0[0] = 1u;                                              // count=1, user mode
  g0[1] = lds_off;                                         // lds_addr
  g0[2] = (unsigned)(ga & 0xFFFFFFFFu);                    // global_addr[31:0]
  g0[3] = (unsigned)((ga >> 32) & 0x01FFFFFFu) | (2u << 30); // addr[56:32] | type=2
  i32x8 g1;
  // data_size=1 (2B) | pad_enable | pad_interval=3 (16 DW) | pad_amount=3 (4 DW)
  g1[0] = (int)((1u << 16) | (1u << 20) | (3u << 22) | (3u << 25));
  g1[1] = (int)((32u & 0xFFFFu) << 16);                    // tensor_dim0[15:0]=32
  g1[2] = (int)((validRows & 0xFFFFu) << 16);              // td0 hi=0 | tensor_dim1 lo
  g1[3] = (int)(((validRows >> 16) & 0xFFFFu) | (32u << 16)); // td1 hi | tile_dim0=32
  g1[4] = (int)(tileRows & 0xFFFFu);                       // tile_dim1 | tile_dim2=0
  g1[5] = (int)strideElems;                                // tensor_dim0_stride[31:0]
  g1[6] = 0;                                               // stride hi | dim1_stride lo
  g1[7] = 0;
  i32x4 z4 = {0, 0, 0, 0};
  i32x8 z8 = {0, 0, 0, 0, 0, 0, 0, 0};
  __builtin_amdgcn_tensor_load_to_lds(g0, g1, z4, z4, z8, 0);
}

// ---------------------------------------------------------------------------
// C[N,BN] = act( concat(A0[N,K0], A1[N,K1]) @ W[BN,K]^T + bias0 + bias1 + res )
// ACT: 0 none, 1 swish, 2 relu.  AF32: A is fp32 (VALU-converted staging),
// else A is bf16 and staged by the Tensor Data Mover.  OUTBF: bf16 output.
// W is bf16 [BN,K] row-major. bias/res are fp32.
// ---------------------------------------------------------------------------
template<int ACT, int BN, bool AF32, bool OUTBF>
__global__ __launch_bounds__(256)
void gemm_bf16_wmma(const void* __restrict__ A0v, int K0,
                    const void* __restrict__ A1v, int K1,
                    const unsigned short* __restrict__ W,
                    const float* __restrict__ bias0,
                    const float* __restrict__ bias1,
                    const float* __restrict__ res, int ldr,
                    void* __restrict__ Cv, int ldc,
                    int nrows)
{
  constexpr int BM = 128, BK = 32, KP = 40;   // KP matches TDM pad: 64B data + 16B pad
  constexpr int NT = BN / 16;

  __shared__ unsigned short sA[BM * KP];
  __shared__ unsigned short sB[BN * KP];

  const int tid  = threadIdx.x;
  const int lane = tid & 31;
  const int wave = tid >> 5;
  const int l15  = lane & 15;
  const int hs   = lane >> 4;
  const int rowBase = blockIdx.x * BM;
  const int K = K0 + K1;

  f32x8 acc[NT] = {};

  for (int k0 = 0; k0 < K; k0 += BK) {
    if (wave == 0) {                       // wave 0 drives the TDM
      if constexpr (!AF32) {
        const unsigned short* A0 = (const unsigned short*)A0v;
        const unsigned short* A1 = (const unsigned short*)A1v;
        const unsigned short* ap;
        unsigned astr;
        if (k0 < K0) { ap = A0 + (size_t)rowBase * K0 + k0;        astr = (unsigned)K0; }
        else         { ap = A1 + (size_t)rowBase * K1 + (k0 - K0); astr = (unsigned)K1; }
        unsigned vrows = (unsigned)(nrows - rowBase);
        if (vrows > (unsigned)BM) vrows = (unsigned)BM;
        tdm_load_tile_bf16((unsigned)(size_t)sA, ap, (unsigned)BM, vrows, astr);
      }
      tdm_load_tile_bf16((unsigned)(size_t)sB, W + (size_t)k0,
                         (unsigned)BN, (unsigned)BN, (unsigned)K);
      __builtin_amdgcn_s_wait_tensorcnt(0);
    }
    if constexpr (AF32) {                  // fp32 A (agg): VALU convert + ds_store
      const float* A0 = (const float*)A0v;
      int r  = tid >> 1;
      int cb = (tid & 1) * 16;
      unsigned int* d32 = (unsigned int*)&sA[r * KP + cb];
      int grow = rowBase + r;
      if (grow < nrows) {
        const float4* s4 = (const float4*)(A0 + (size_t)grow * K + k0 + cb);
#pragma unroll
        for (int c = 0; c < 4; ++c) {
          float4 v = s4[c];
          d32[2 * c]     = pack_bf16x2(v.x, v.y);
          d32[2 * c + 1] = pack_bf16x2(v.z, v.w);
        }
      } else {
#pragma unroll
        for (int c = 0; c < 8; ++c) d32[c] = 0u;
      }
    }
    __syncthreads();

    // A fragment (16x32 bf16), k = 16*hs + s per lane half, same order as B
    union { bf16x16 v; unsigned int u[8]; } afrag;
    {
      const unsigned short* ab = &sA[(wave * 16 + l15) * KP + hs * 16];
#pragma unroll
      for (int p = 0; p < 4; ++p) {
        afrag.u[p]     = *(const unsigned int*)(ab + 2 * p);
        afrag.u[4 + p] = *(const unsigned int*)(ab + 8 + 2 * p);
      }
    }
#pragma unroll
    for (int j = 0; j < NT; ++j) {
      union { bf16x16 v; unsigned int u[8]; } bfrag;
      const unsigned short* bb = &sB[(j * 16 + l15) * KP + hs * 16];
#pragma unroll
      for (int p = 0; p < 8; ++p)
        bfrag.u[p] = *(const unsigned int*)(bb + 2 * p);
      acc[j] = __builtin_amdgcn_wmma_f32_16x16x32_bf16(
          false, afrag.v, false, bfrag.v, (short)0, acc[j], false, false);
    }
    __syncthreads();
  }

  // epilogue
#pragma unroll
  for (int j = 0; j < NT; ++j) {
    int col = j * 16 + l15;
    float badd = 0.0f;
    if (bias0) badd += bias0[col];
    if (bias1) badd += bias1[col];
#pragma unroll
    for (int i = 0; i < 8; ++i) {
      int row = rowBase + wave * 16 + i + hs * 8;
      if (row < nrows) {
        float v = acc[j][i] + badd;
        if (res) v += res[(size_t)row * ldr + col];
        if (ACT == 1)      v = v * (1.0f / (1.0f + __expf(-v)));
        else if (ACT == 2) v = v > 0.0f ? v : 0.0f;
        if constexpr (OUTBF)
          ((unsigned short*)Cv)[(size_t)row * ldc + col] = f32_bf16_bits(v);
        else
          ((float*)Cv)[(size_t)row * ldc + col] = v;
      }
    }
  }
}

// ---------- small utility kernels ----------
__global__ void f32_to_bf16_kernel(const float* __restrict__ in,
                                   unsigned short* __restrict__ out, long long n) {
  long long i = (long long)blockIdx.x * blockDim.x + threadIdx.x;
  if (i < n) out[i] = f32_bf16_bits(in[i]);
}

__global__ void fill_zero_f32(float* __restrict__ p, long long n) {
  long long i = (long long)blockIdx.x * blockDim.x + threadIdx.x;
  if (i < n) p[i] = 0.0f;
}

__global__ void deg_count_kernel(const int* __restrict__ dst,
                                 const int* __restrict__ et,
                                 float* __restrict__ deg, int E) {
  int e = blockIdx.x * blockDim.x + threadIdx.x;
  if (e < E) unsafeAtomicAdd(&deg[dst[e] * NREL + et[e]], 1.0f);
}

__global__ void deg_inv_kernel(float* __restrict__ deg, int n) {
  int i = blockIdx.x * blockDim.x + threadIdx.x;
  if (i < n) {
    float d = deg[i];
    deg[i] = d > 0.0f ? 1.0f / d : 0.0f;
  }
}

// per-(dst,rel) mean scatter from bf16 features into fp32 agg; 32 threads/edge
__global__ void scatter_mean_kernel(const unsigned short* __restrict__ x,
                                    const int* __restrict__ src,
                                    const int* __restrict__ dst,
                                    const int* __restrict__ et,
                                    const float* __restrict__ invdeg,
                                    float* __restrict__ agg, int E) {
  int gid = blockIdx.x * blockDim.x + threadIdx.x;
  int e = gid >> 5;
  if (e >= E) return;
  int part = gid & 31;
  int seg = dst[e] * NREL + et[e];
  float w = invdeg[seg];
  const unsigned int* xp = (const unsigned int*)(x + (size_t)src[e] * HDIM + part * 4);
  unsigned int a = xp[0], b = xp[1];
  float* out = agg + (size_t)seg * HDIM + part * 4;
  unsafeAtomicAdd(out + 0, bf16lo_f32(a) * w);
  unsafeAtomicAdd(out + 1, bf16hi_f32(a) * w);
  unsafeAtomicAdd(out + 2, bf16lo_f32(b) * w);
  unsafeAtomicAdd(out + 3, bf16hi_f32(b) * w);
}

__global__ void add_inplace_bf16(unsigned short* __restrict__ a,
                                 const unsigned short* __restrict__ b, long long n) {
  long long i = (long long)blockIdx.x * blockDim.x + threadIdx.x;
  if (i < n) {
    float va, vb;
    { union { unsigned int u; float f; } c; c.u = (unsigned int)a[i] << 16; va = c.f; }
    { union { unsigned int u; float f; } c; c.u = (unsigned int)b[i] << 16; vb = c.f; }
    a[i] = f32_bf16_bits(va + vb);
  }
}

extern "C" void kernel_launch(void* const* d_in, const int* in_sizes, int n_in,
                              void* d_out, int out_size, void* d_ws, size_t ws_size,
                              hipStream_t stream)
{
  (void)in_sizes; (void)n_in; (void)out_size; (void)ws_size;
  const int N = NNODES, E = NEDGES;

  const float* mono = (const float*)d_in[0];
  const float* atom = (const float*)d_in[1];
  const int*   ei   = (const int*)d_in[2];
  const int*   et   = (const int*)d_in[3];
  const int*   esrc = ei;
  const int*   edst = ei + E;

  int p = 4;
  const float* lina1_w = (const float*)d_in[p++]; const float* lina1_b = (const float*)d_in[p++];
  const float* lina2_w = (const float*)d_in[p++]; const float* lina2_b = (const float*)d_in[p++];
  const float* f0_w1   = (const float*)d_in[p++]; const float* f0_w2   = (const float*)d_in[p++];
  const float* f1_w1   = (const float*)d_in[p++]; const float* f1_w2   = (const float*)d_in[p++];
  const float* c0_loop_w = (const float*)d_in[p++]; const float* c0_loop_b = (const float*)d_in[p++];
  const float* c0_rel_w  = (const float*)d_in[p++]; const float* c0_rel_b  = (const float*)d_in[p++];
  const float* c1_loop_w = (const float*)d_in[p++]; const float* c1_loop_b = (const float*)d_in[p++];
  const float* c1_rel_w  = (const float*)d_in[p++]; const float* c1_rel_b  = (const float*)d_in[p++];
  const float* linp0_w = (const float*)d_in[p++]; const float* linp0_b = (const float*)d_in[p++];
  const float* linp1_w = (const float*)d_in[p++]; const float* linp1_b = (const float*)d_in[p++];
  const float* cat_w0 = (const float*)d_in[p++];
  const float* cat_w1 = (const float*)d_in[p++];
  const float* cat_w2 = (const float*)d_in[p++];
  const float* cat_b0 = (const float*)d_in[p++];
  const float* cat_b1 = (const float*)d_in[p++];
  const float* cat_b2 = (const float*)d_in[p++];
  const float* lins_w0 = (const float*)d_in[p++];
  const float* lins_w1 = (const float*)d_in[p++];
  const float* lins_b0 = (const float*)d_in[p++];
  const float* lins_b1 = (const float*)d_in[p++];
  const float* final_w = (const float*)d_in[p++];
  const float* final_b = (const float*)d_in[p++];

  // workspace: fp32 region first, then bf16 region (256B aligned chunks)
  char* wsb = (char*)d_ws;
  size_t off = 0;
  auto allocf = [&](size_t n) -> float* {
    float* r = (float*)(wsb + off); off = (off + n * 4 + 255) & ~(size_t)255; return r;
  };
  auto alloch = [&](size_t n) -> unsigned short* {
    unsigned short* r = (unsigned short*)(wsb + off); off = (off + n * 2 + 255) & ~(size_t)255; return r;
  };
  float* preb = allocf((size_t)N * HDIM);
  float* agg  = allocf((size_t)N * NREL * HDIM);
  float* deg  = allocf((size_t)N * NREL);
  unsigned short* mono16 = alloch((size_t)N * HDIM);
  unsigned short* atom16 = alloch((size_t)N * ADIM);
  unsigned short* x1   = alloch((size_t)N * HDIM);
  unsigned short* x2   = alloch((size_t)N * HDIM);
  unsigned short* fbuf = alloch((size_t)N * HDIM);
  unsigned short* midb = alloch((size_t)N * MIDDIM);
  unsigned short* hbuf = alloch((size_t)N * HDIM);
  unsigned short* hcat = alloch((size_t)N * 2 * HDIM);
  // bf16 weights
  unsigned short* w_lina1 = alloch(128 * 128);
  unsigned short* w_lina2 = alloch(128 * 128);
  unsigned short* w_f0w1  = alloch(64 * 128);
  unsigned short* w_f0w2  = alloch(128 * 64);
  unsigned short* w_f1w1  = alloch(64 * 192);
  unsigned short* w_f1w2  = alloch(128 * 64);
  unsigned short* w_c0l   = alloch(128 * 128);
  unsigned short* w_c0r   = alloch(128 * 512);
  unsigned short* w_c1l   = alloch(128 * 128);
  unsigned short* w_c1r   = alloch(128 * 512);
  unsigned short* w_lp0   = alloch(128 * 128);
  unsigned short* w_lp1   = alloch(128 * 128);
  unsigned short* w_cat0  = alloch(128 * 256);
  unsigned short* w_cat1  = alloch(128 * 128);
  unsigned short* w_cat2  = alloch(128 * 128);
  unsigned short* w_ls0   = alloch(128 * 128);
  unsigned short* w_ls1   = alloch(128 * 128);
  unsigned short* w_fin   = alloch(128 * 128);

  dim3 blk(256);
  dim3 gM((N + 127) / 128);
  long long nd   = (long long)N * NREL;
  long long nagg = (long long)N * NREL * HDIM;
  long long nh   = (long long)N * HDIM;
  unsigned degBlocks  = (unsigned)((nd + 255) / 256);
  unsigned aggBlocks  = (unsigned)((nagg + 255) / 256);
  unsigned addBlocks  = (unsigned)((nh + 255) / 256);
  unsigned scatBlocks = (unsigned)(((long long)E * 32 + 255) / 256);

  auto cvt = [&](const float* s, unsigned short* d, long long n) {
    f32_to_bf16_kernel<<<(unsigned)((n + 255) / 256), blk, 0, stream>>>(s, d, n);
  };

  // one-time per launch: inputs + weights to bf16
  cvt(mono, mono16, (long long)N * HDIM);
  cvt(atom, atom16, (long long)N * ADIM);
  cvt(lina1_w, w_lina1, 128 * 128);  cvt(lina2_w, w_lina2, 128 * 128);
  cvt(f0_w1, w_f0w1, 64 * 128);      cvt(f0_w2, w_f0w2, 128 * 64);
  cvt(f1_w1, w_f1w1, 64 * 192);      cvt(f1_w2, w_f1w2, 128 * 64);
  cvt(c0_loop_w, w_c0l, 128 * 128);  cvt(c0_rel_w, w_c0r, 128 * 512);
  cvt(c1_loop_w, w_c1l, 128 * 128);  cvt(c1_rel_w, w_c1r, 128 * 512);
  cvt(linp0_w, w_lp0, 128 * 128);    cvt(linp1_w, w_lp1, 128 * 128);
  cvt(cat_w0, w_cat0, 128 * 256);    cvt(cat_w1, w_cat1, 128 * 128);
  cvt(cat_w2, w_cat2, 128 * 128);
  cvt(lins_w0, w_ls0, 128 * 128);    cvt(lins_w1, w_ls1, 128 * 128);
  cvt(final_w, w_fin, 128 * 128);

  // degrees (feature independent)
  fill_zero_f32<<<degBlocks, blk, 0, stream>>>(deg, nd);
  deg_count_kernel<<<(E + 255) / 256, blk, 0, stream>>>(edst, et, deg, E);
  deg_inv_kernel<<<degBlocks, blk, 0, stream>>>(deg, (int)nd);

  // x1 / x2
  gemm_bf16_wmma<1,128,false,true><<<gM, blk, 0, stream>>>(mono16,128, nullptr,0, w_lina1, lina1_b, nullptr, nullptr,0, x1,128, N);
  gemm_bf16_wmma<1,128,false,true><<<gM, blk, 0, stream>>>(mono16,128, nullptr,0, w_lina2, lina2_b, nullptr, nullptr,0, x2,128, N);

  // f0
  gemm_bf16_wmma<0, 64,false,true><<<gM, blk, 0, stream>>>(x1,128, nullptr,0, w_f0w1, nullptr, nullptr, nullptr,0, midb,64, N);
  gemm_bf16_wmma<0,128,false,true><<<gM, blk, 0, stream>>>(midb,64, nullptr,0, w_f0w2, nullptr, nullptr, nullptr,0, fbuf,128, N);

  // conv0
  fill_zero_f32<<<aggBlocks, blk, 0, stream>>>(agg, nagg);
  scatter_mean_kernel<<<scatBlocks, blk, 0, stream>>>(fbuf, esrc, edst, et, deg, agg, E);
  gemm_bf16_wmma<0,128,false,false><<<gM, blk, 0, stream>>>(fbuf,128, nullptr,0, w_c0l, c0_loop_b, c0_rel_b, nullptr,0, preb,128, N);
  gemm_bf16_wmma<2,128,true ,true ><<<gM, blk, 0, stream>>>(agg,512, nullptr,0, w_c0r, nullptr, nullptr, preb,128, hbuf,128, N);
  gemm_bf16_wmma<1,128,false,true><<<gM, blk, 0, stream>>>(hbuf,128, nullptr,0, w_lp0, linp0_b, nullptr, nullptr,0, hcat,256, N);

  // f1
  gemm_bf16_wmma<0, 64,false,true><<<gM, blk, 0, stream>>>(x1,128, atom16,ADIM, w_f1w1, nullptr, nullptr, nullptr,0, midb,64, N);
  gemm_bf16_wmma<0,128,false,true><<<gM, blk, 0, stream>>>(midb,64, nullptr,0, w_f1w2, nullptr, nullptr, nullptr,0, fbuf,128, N);

  // conv1
  fill_zero_f32<<<aggBlocks, blk, 0, stream>>>(agg, nagg);
  scatter_mean_kernel<<<scatBlocks, blk, 0, stream>>>(fbuf, esrc, edst, et, deg, agg, E);
  gemm_bf16_wmma<0,128,false,false><<<gM, blk, 0, stream>>>(fbuf,128, nullptr,0, w_c1l, c1_loop_b, c1_rel_b, nullptr,0, preb,128, N);
  gemm_bf16_wmma<2,128,true ,true ><<<gM, blk, 0, stream>>>(agg,512, nullptr,0, w_c1r, nullptr, nullptr, preb,128, hbuf,128, N);
  gemm_bf16_wmma<1,128,false,true><<<gM, blk, 0, stream>>>(hbuf,128, nullptr,0, w_lp1, linp1_b, nullptr, nullptr,0, hcat + HDIM,256, N);

  // cat MLP (bf16 ping-pong: hbuf -> fbuf -> hbuf)
  gemm_bf16_wmma<1,128,false,true><<<gM, blk, 0, stream>>>(hcat,256, nullptr,0, w_cat0, cat_b0, nullptr, nullptr,0, hbuf,128, N);
  gemm_bf16_wmma<1,128,false,true><<<gM, blk, 0, stream>>>(hbuf,128, nullptr,0, w_cat1, cat_b1, nullptr, nullptr,0, fbuf,128, N);
  gemm_bf16_wmma<1,128,false,true><<<gM, blk, 0, stream>>>(fbuf,128, nullptr,0, w_cat2, cat_b2, nullptr, nullptr,0, hbuf,128, N);

  // h = h + x2
  add_inplace_bf16<<<addBlocks, blk, 0, stream>>>(hbuf, x2, nh);

  // lins
  gemm_bf16_wmma<1,128,false,true><<<gM, blk, 0, stream>>>(hbuf,128, nullptr,0, w_ls0, lins_b0, nullptr, nullptr,0, fbuf,128, N);
  gemm_bf16_wmma<1,128,false,true><<<gM, blk, 0, stream>>>(fbuf,128, nullptr,0, w_ls1, lins_b1, nullptr, nullptr,0, hbuf,128, N);

  // final (fp32 out)
  gemm_bf16_wmma<0,128,false,false><<<gM, blk, 0, stream>>>(hbuf,128, nullptr,0, w_fin, final_b, nullptr, nullptr,0, (float*)d_out,128, N);
}
